// Net_82764019793920
// MI455X (gfx1250) — matrix operand order repaired and checked by reading, not compiled, not run on hardware
//
// CDNA5 (gfx1250) implementation of the 5-branch edge-conditioned GCN stack.
//
// concat([x_i, x_j, ea]) @ We == (x@We1)[row] + (x@We2)[col] + ea@We3, so all
// gather-GEMMs become node-space GEMMs + L2-resident per-edge gathers; only
// ea@We3 stays edge-space. segment_sum fused into the edge-GEMM epilogue via
// global f32 atomics. Inter-layer edge features ("m") stored bf16 with rows
// padded to 64 (zero-filled) so the next layer's A-loads are branch-free.
//
// v_wmma_f32_16x16x32_bf16 (wave32); B staged in LDS pre-swizzled to fragment
// order; 4 N-tiles computed unconditionally (B zero-padded); K-loop split into
// a branch-free fast part + general tail; OOB row strips handled by wave-level
// early exit (M is always a multiple of 16).

#include <hip/hip_runtime.h>
#include <hip/hip_bf16.h>

typedef __attribute__((ext_vector_type(16))) __bf16 v16bf;
typedef __attribute__((ext_vector_type(8)))  float  v8f;

#define NNODES 50000
#define NEDGES 500000
#define HDIM   50
#define MPAD   64          // padded row length of bf16 edge-feature matrix m
#define BN_EPS 1e-5f

static __device__ __forceinline__ unsigned short f2bfu(float f) {
  union { __bf16 h; unsigned short u; } z;
  z.h = (__bf16)f;                       // hardware RNE (v_cvt_*bf16_f32)
  return z.u;
}
static __device__ __forceinline__ float bf2f(unsigned short s) {
  union { unsigned u; float f; } x; x.u = ((unsigned)s) << 16;
  return x.f;
}
static __host__ __device__ __forceinline__ int imin(int a, int b) { return a < b ? a : b; }

// ---------------------------------------------------------------------------
// C[M x Nout] = [A1 | A2] @ B (+bias), fp32 accum via bf16 WMMA.
//   A1: M x K1 (fp32 or bf16, leading dim lda1), A2: M x K2 fp32 (optional).
//   kfast: k0 < kfast -> all lanes' 8-wide A1 loads are in-bounds (branch-free).
// mode 0: C[gr*ldc + ccol + gc] = v (+bias)
// mode 1: (edge layer, Nout==HDIM) v += P1[row[e]]+P2[col[e]];
//         atomicAdd(agg[row[e]]); mOut[e*MPAD+gc] = bf16(v), pad cols zeroed.
// Block = 256 threads = 8 waves; wave = 16 rows x 64 cols; grid.x: M/128,
// grid.y: Nout/64.  Requires M % 16 == 0.
// ---------------------------------------------------------------------------
__global__ __launch_bounds__(256) void gemm_wmma(
    const void* __restrict__ A1, int lda1, int K1, int a1bf, int kfast,
    const float* __restrict__ A2, int lda2, int K2,
    const float* __restrict__ B, int ldb,
    const float* __restrict__ bias,
    int M, int Nout, int mode,
    float* __restrict__ C, int ldc, int ccol,
    const int* __restrict__ erow, const int* __restrict__ ecol,
    const float* __restrict__ P1, const float* __restrict__ P2,
    float* __restrict__ agg, unsigned short* __restrict__ mOut)
{
  // B in LDS, bf16, fragment order:
  // element ((kc*4 + t)*32 + lane)*16 + i  <->  B[kc*32 + (lane<16?0:8) + i + (i<8?0:8)][t*16 + lane%16]
  __shared__ __align__(32) unsigned short Bs[352 * 64];

  const int tid   = threadIdx.x;
  const int lane  = tid & 31;
  const int wave  = tid >> 5;
  const int Krows = K1 + K2;
  const int Kpad  = (Krows + 31) & ~31;          // <= 352 for all layers here
  const int ng    = blockIdx.y;
  const int ncols = imin(64, Nout - ng * 64);
  const int ntile = (ncols + 15) >> 4;

  const int tot = Kpad * 64;
  for (int d = tid; d < tot; d += 256) {
    int i  = d & 15;
    int ln = (d >> 4) & 31;
    int t  = (d >> 9) & 3;
    int kc = d >> 11;
    int k  = kc * 32 + ((ln < 16) ? 0 : 8) + i + ((i < 8) ? 0 : 8);
    int c  = ng * 64 + t * 16 + (ln & 15);
    float v = 0.f;
    if (k < Krows && c < Nout) v = B[(size_t)k * ldb + c];
    Bs[d] = f2bfu(v);
  }
  __syncthreads();

  const int m0 = blockIdx.x * 128 + wave * 16;
  if (m0 >= M) return;                 // whole strip OOB (M % 16 == 0); safe: no barriers below

  const int r    = m0 + (lane & 15);
  const int aoff = (lane < 16) ? 0 : 8;          // per-lane K sub-block (A layout)
  const unsigned short* ldsbase = &Bs[lane * 16];

  v8f zz = {};
  v8f acc[4];
  acc[0] = zz; acc[1] = zz; acc[2] = zz; acc[3] = zz;

  union AFrag { v16bf v; unsigned short u[16]; __bf16 h[16]; };

#define WMMA_STEP(AFV, K0)                                                     \
  do {                                                                         \
    const unsigned short* lb = ldsbase + ((K0) >> 5) * 2048;                   \
    v16bf b0 = *(const v16bf*)(lb);                                            \
    v16bf b1 = *(const v16bf*)(lb + 512);                                      \
    v16bf b2 = *(const v16bf*)(lb + 1024);                                     \
    v16bf b3 = *(const v16bf*)(lb + 1536);                                     \
    acc[0] = __builtin_amdgcn_wmma_f32_16x16x32_bf16(false, (AFV), false, b0,  \
                                                     (short)0, acc[0], false, false); \
    acc[1] = __builtin_amdgcn_wmma_f32_16x16x32_bf16(false, (AFV), false, b1,  \
                                                     (short)0, acc[1], false, false); \
    acc[2] = __builtin_amdgcn_wmma_f32_16x16x32_bf16(false, (AFV), false, b2,  \
                                                     (short)0, acc[2], false, false); \
    acc[3] = __builtin_amdgcn_wmma_f32_16x16x32_bf16(false, (AFV), false, b3,  \
                                                     (short)0, acc[3], false, false); \
  } while (0)

  int k0 = 0;
  if (a1bf) {
    const unsigned short* arow = (const unsigned short*)A1 + (size_t)r * lda1;
    for (; k0 < kfast; k0 += 32) {
      AFrag af;
      const unsigned* p0 = (const unsigned*)(arow + k0 + aoff);        // K k0+aoff..+7
      const unsigned* p1 = (const unsigned*)(arow + k0 + 16 + aoff);   // K k0+16+aoff..+7
      unsigned* d0 = (unsigned*)&af.u[0];
      unsigned* d1 = (unsigned*)&af.u[8];
#pragma unroll
      for (int i = 0; i < 4; ++i) { d0[i] = p0[i]; d1[i] = p1[i]; }
      WMMA_STEP(af.v, k0);
    }
  } else {
    const float* arow = (const float*)A1 + (size_t)r * lda1;
    for (; k0 < kfast; k0 += 32) {
      AFrag af;
      const float* p0 = arow + k0 + aoff;
      const float* p1 = arow + k0 + 16 + aoff;
#pragma unroll
      for (int i = 0; i < 8; ++i) {
        af.h[i]     = (__bf16)p0[i];
        af.h[8 + i] = (__bf16)p1[i];
      }
      WMMA_STEP(af.v, k0);
    }
  }
  // general tail (K1 boundary / A2 region / zero padding)
  for (; k0 < Kpad; k0 += 32) {
    AFrag af;
#pragma unroll
    for (int g = 0; g < 2; ++g) {
      const int kk = k0 + g * 16 + aoff;
#pragma unroll
      for (int i = 0; i < 8; ++i) {
        int k = kk + i; float v = 0.f;
        if (k < K1)
          v = a1bf ? bf2f(((const unsigned short*)A1)[(size_t)r * lda1 + k])
                   : ((const float*)A1)[(size_t)r * lda1 + k];
        else if (k < Krows)
          v = A2[(size_t)r * lda2 + (k - K1)];
        af.h[g * 8 + i] = (__bf16)v;
      }
    }
    WMMA_STEP(af.v, k0);
  }
#undef WMMA_STEP

  // Epilogue. C layout: element j -> row m0 + j + (lane>=16 ? 8:0), col lane%16.
  const int lrb = (lane >= 16) ? 8 : 0;
  if (mode == 0) {
#pragma unroll
    for (int t = 0; t < 4; ++t) {
      if (t >= ntile) break;
      int gc = ng * 64 + t * 16 + (lane & 15);
      if (gc >= Nout) continue;
      float bv = bias ? bias[gc] : 0.f;
#pragma unroll
      for (int j = 0; j < 8; ++j) {
        int gr = m0 + lrb + j;
        C[(size_t)gr * ldc + ccol + gc] = acc[t][j] + bv;
      }
    }
  } else {
    // mode 1: Nout == HDIM (50); tiles cover cols 0..63 == padded m row
#pragma unroll
    for (int t = 0; t < 4; ++t) {
      int gc = t * 16 + (lane & 15);
#pragma unroll
      for (int j = 0; j < 8; ++j) {
        int e = m0 + lrb + j;
        if (gc < HDIM) {
          int rn = erow[e], cn = ecol[e];
          float v = acc[t][j] + P1[(size_t)rn * HDIM + gc] + P2[(size_t)cn * HDIM + gc];
          atomicAdd(&agg[(size_t)rn * HDIM + gc], v);    // segment_sum
          mOut[(size_t)e * MPAD + gc] = f2bfu(v);        // ea for next layer
        } else {
          mOut[(size_t)e * MPAD + gc] = 0;               // keep pad finite
        }
      }
    }
  }
}

// ---------------------------------------------------------------------------
__global__ __launch_bounds__(256) void zerof(float* p, int n) {
  int i = blockIdx.x * 256 + threadIdx.x;
  if (i < n) p[i] = 0.f;
}

__global__ __launch_bounds__(256) void concat_h(const float* __restrict__ x,
                                                const float* __restrict__ c,
                                                float* __restrict__ h, int n) {
  int i = blockIdx.x * 256 + threadIdx.x;
  if (i >= n * 46) return;
  int r = i / 46, j = i % 46;
  h[i] = (j < 26) ? x[(size_t)r * 26 + j] : c[(size_t)r * 20 + (j - 26)];
}

// Per-channel sum / sumsq via LDS atomics, flushed once per block to global.
__global__ __launch_bounds__(256) void bn_stats(const float* __restrict__ X,
                                                int M, int C,
                                                float* __restrict__ stats) {
  __shared__ float s[196];
  __shared__ float q[196];
  int tid = threadIdx.x;
  for (int i = tid; i < C; i += 256) { s[i] = 0.f; q[i] = 0.f; }
  __syncthreads();
  long long tot = (long long)M * C;
  for (long long idx = (long long)blockIdx.x * 256 + tid; idx < tot;
       idx += (long long)gridDim.x * 256) {
    float v = X[idx];
    int   c = (int)(idx % C);
    atomicAdd(&s[c], v);
    atomicAdd(&q[c], v * v);
  }
  __syncthreads();
  for (int i = tid; i < C; i += 256) {
    atomicAdd(&stats[i],     s[i]);
    atomicAdd(&stats[C + i], q[i]);
  }
}

__global__ __launch_bounds__(256) void bn_apply(const float* __restrict__ X,
                                                int M, int C,
                                                const float* __restrict__ stats,
                                                const float* __restrict__ g,
                                                const float* __restrict__ b,
                                                float* __restrict__ dst,
                                                int dld, int dcol, int elu) {
  long long idx = (long long)blockIdx.x * 256 + threadIdx.x;
  if (idx >= (long long)M * C) return;
  int r = (int)(idx / C), c = (int)(idx % C);
  float mean = stats[c] / (float)M;
  float var  = stats[C + c] / (float)M - mean * mean;
  float v = (X[idx] - mean) * rsqrtf(var + BN_EPS) * g[c] + b[c];
  if (elu && v <= 0.f) v = expm1f(v);
  dst[(size_t)r * dld + dcol + c] = v;
}

// ---------------------------------------------------------------------------
static inline void launch_gemm(hipStream_t stream,
    const void* A1, int lda1, int K1, int a1bf, int kfast,
    const float* A2, int lda2, int K2,
    const float* B, int ldb, const float* bias,
    int M, int Nout, int mode,
    float* C, int ldc, int ccol,
    const int* erow, const int* ecol,
    const float* P1, const float* P2, float* agg, unsigned short* mOut)
{
  dim3 grid((M + 127) / 128, (Nout + 63) / 64);
  gemm_wmma<<<grid, 256, 0, stream>>>(A1, lda1, K1, a1bf, kfast, A2, lda2, K2,
                                      B, ldb, bias, M, Nout, mode, C, ldc, ccol,
                                      erow, ecol, P1, P2, agg, mOut);
}

extern "C" void kernel_launch(void* const* d_in, const int* in_sizes, int n_in,
                              void* d_out, int out_size, void* d_ws, size_t ws_size,
                              hipStream_t stream) {
  (void)in_sizes; (void)n_in; (void)out_size; (void)ws_size;

  // setup_inputs() insertion order, then params flattened in JAX pytree order
  // (dict keys sorted: WlinR, bR, blinR, bns(list:[b,g] per layer), convs(list:
  //  [We,Wn,be,bn] per layer), gR).
  const float* x     = (const float*)d_in[0];
  const int*   ei    = (const int*)d_in[1];
  const float* ea0   = (const float*)d_in[2];
  const float* cond  = (const float*)d_in[3];
  const float* WlinR = (const float*)d_in[4];
  const float* bR    = (const float*)d_in[5];
  const float* blinR = (const float*)d_in[6];
  const float* gR    = (const float*)d_in[97];
  const int* erow = ei;
  const int* ecol = ei + NEDGES;

  // --- workspace carve-up (~230 MB) ---
  char* ws = (char*)d_ws;
  auto carve = [&](size_t bytes) { char* p = ws; ws += (bytes + 255) & ~(size_t)255; return p; };
  float*          h     = (float*)carve((size_t)NNODES * 46 * 4);
  float*          xi    = (float*)carve((size_t)NNODES * HDIM * 4);
  float*          xn    = (float*)carve((size_t)NNODES * HDIM * 4);
  float*          P1    = (float*)carve((size_t)NNODES * HDIM * 4);
  float*          P2    = (float*)carve((size_t)NNODES * HDIM * 4);
  float*          agg   = (float*)carve((size_t)NNODES * HDIM * 4);
  float*          o     = (float*)carve((size_t)NNODES * 250 * 4);
  unsigned short* mA    = (unsigned short*)carve((size_t)NEDGES * MPAD * 2);
  unsigned short* mB    = (unsigned short*)carve((size_t)NEDGES * MPAD * 2);
  float*          stats = (float*)carve(512 * 4);

  concat_h<<<(NNODES * 46 + 255) / 256, 256, 0, stream>>>(x, cond, h, NNODES);

  for (int b = 0; b < 5; ++b) {
    const float*    curx  = h;    int cin = 46;
    const void*     curEa = ea0;  int ce  = 345; int ceLd = 345;
    int             eaBF  = 0;    int ceKfast = (345 / 32) * 32;   // 320
    unsigned short* mcur  = mA;
    for (int l = 0; l <= b; ++l) {
      int L = b * (b + 1) / 2 + l;
      const float* We  = (const float*)d_in[37 + 4 * L + 0];  // (2cin+ce) x 50
      const float* Wn  = (const float*)d_in[37 + 4 * L + 1];  // (cin+50) x 50
      const float* be  = (const float*)d_in[37 + 4 * L + 2];
      const float* bnb = (const float*)d_in[37 + 4 * L + 3];
      const float* bnB = (const float*)d_in[7 + 2 * L + 0];   // BN shift
      const float* bnG = (const float*)d_in[7 + 2 * L + 1];   // BN scale
      int xkfast = (cin / 32) * 32;

      // P1 = curx @ We[0:cin] + be ; P2 = curx @ We[cin:2cin]
      launch_gemm(stream, curx, cin, cin, 0, xkfast, nullptr, 0, 0, We, HDIM, be,
                  NNODES, HDIM, 0, P1, HDIM, 0, nullptr, nullptr, nullptr, nullptr, nullptr, nullptr);
      launch_gemm(stream, curx, cin, cin, 0, xkfast, nullptr, 0, 0,
                  We + (size_t)cin * HDIM, HDIM, nullptr,
                  NNODES, HDIM, 0, P2, HDIM, 0, nullptr, nullptr, nullptr, nullptr, nullptr, nullptr);

      zerof<<<(NNODES * HDIM + 255) / 256, 256, 0, stream>>>(agg, NNODES * HDIM);

      // edge: m = ea @ We[2cin:] + P1[row] + P2[col]; agg[row] += m; store bf16 m (lda MPAD)
      launch_gemm(stream, curEa, ceLd, ce, eaBF, ceKfast, nullptr, 0, 0,
                  We + (size_t)2 * cin * HDIM, HDIM, nullptr,
                  NEDGES, HDIM, 1, nullptr, 0, 0, erow, ecol, P1, P2, agg, mcur);

      // xn = [curx | agg] @ Wn + bn
      launch_gemm(stream, curx, cin, cin, 0, xkfast, agg, HDIM, HDIM, Wn, HDIM, bnb,
                  NNODES, HDIM, 0, xn, HDIM, 0, nullptr, nullptr, nullptr, nullptr, nullptr, nullptr);

      // BN (batch stats) + ELU
      zerof<<<(512 + 255) / 256, 256, 0, stream>>>(stats, 512);
      bn_stats<<<512, 256, 0, stream>>>(xn, NNODES, HDIM, stats);
      if (l == b)   // branch output -> o[:, b*50 : b*50+50]
        bn_apply<<<(NNODES * HDIM + 255) / 256, 256, 0, stream>>>(
            xn, NNODES, HDIM, stats, bnG, bnB, o, 250, b * HDIM, 1);
      else
        bn_apply<<<(NNODES * HDIM + 255) / 256, 256, 0, stream>>>(
            xn, NNODES, HDIM, stats, bnG, bnB, xi, HDIM, 0, 1);

      curx = xi; cin = HDIM;
      curEa = mcur; ce = HDIM; ceLd = MPAD; eaBF = 1; ceKfast = MPAD;  // fully branch-free
      mcur = (mcur == mA) ? mB : mA;
    }
  }

  // final: d_out = BN(o @ WlinR + blinR) ; reshape is a no-op
  float* outF = (float*)d_out;
  launch_gemm(stream, o, 250, 250, 0, (250 / 32) * 32, nullptr, 0, 0, WlinR, 196, blinR,
              NNODES, 196, 0, outF, 196, 0, nullptr, nullptr, nullptr, nullptr, nullptr, nullptr);
  zerof<<<(512 + 255) / 256, 256, 0, stream>>>(stats, 512);
  bn_stats<<<512, 256, 0, stream>>>(outF, NNODES, 196, stats);
  bn_apply<<<(NNODES * 196 + 255) / 256, 256, 0, stream>>>(
      outF, NNODES, 196, stats, gR, bR, outF, 196, 0, 0);
}
